// CoNystromAttention_73581379715668
// MI455X (gfx1250) — matrix operand order repaired and checked by reading, not compiled, not run on hardware
//
#include <hip/hip_runtime.h>
#include <hip/hip_bf16.h>

typedef __attribute__((ext_vector_type(16))) _Float16 v16h;
typedef __attribute__((ext_vector_type(8)))  float    v8f;

#define EMBED 512
#define NHEAD 8
#define HDIM  64
#define NLM   8
#define WIN   64
#define SEQ   1024
#define NBATCH 16
#define INV_SCALE 0.35355339059327373f   // 64^-0.25

#ifndef __has_builtin
#define __has_builtin(x) 0
#endif
#if __has_builtin(__builtin_amdgcn_global_load_async_to_lds_b32) && \
    __has_builtin(__builtin_amdgcn_s_wait_asynccnt)
#define HAVE_ASYNC 1
#else
#define HAVE_ASYNC 0
#endif

// The async builtins take address-space-qualified int pointers.
#define AS1G(p) ((__attribute__((address_space(1))) int*)((void*)(p)))
#define AS3L(p) ((__attribute__((address_space(3))) int*)((void*)(p)))

__device__ __forceinline__ float dot64(const float* a, const float* b) {
  float s = 0.0f;
#pragma unroll
  for (int e = 0; e < 64; ++e) s = fmaf(a[e], b[e], s);
  return s;
}

// Build a 16-wide f16 fragment from two contiguous 8-float LDS runs.
__device__ __forceinline__ v16h frag_from_f32(const float* p0, const float* p1) {
  v16h f;
#pragma unroll
  for (int e = 0; e < 8; ++e) {
    f[e]     = (_Float16)p0[e];
    f[e + 8] = (_Float16)p1[e];
  }
  return f;
}

// ---------------------------------------------------------------------------
// Kernel 1: QKV projection, WMMA f16->f32, async-LDS staged, double buffered.
// Block = 128 threads (4 waves) computes a 64(s) x 64(d) tile of one of
// {Q,K,V} for one (b,h).  Per K-step a 32x64 X-slab and 32x64 W-slab are
// staged (transposed) into LDS by the async global->LDS engine and shared by
// all 4 waves; each wave runs 4 WMMAs per step on its 16-row stripe.
// ---------------------------------------------------------------------------
__device__ __forceinline__ void stage_slab(int tid, const float* gA, const float* gW,
                                           float (*At)[36], float (*Wt)[36]) {
#pragma unroll
  for (int j = 0; j < 16; ++j) {
    const int tt = tid + j * 128;            // 0..2047
    const int k = tt >> 6, c = tt & 63;      // k: 0..31, c: s or n
#if HAVE_ASYNC
    __builtin_amdgcn_global_load_async_to_lds_b32(
        AS1G(gA + (size_t)k * SEQ + c), AS3L(&At[c][k]), 0, 0);
    __builtin_amdgcn_global_load_async_to_lds_b32(
        AS1G(gW + (size_t)k * HDIM + c), AS3L(&Wt[c][k]), 0, 0);
#else
    At[c][k] = gA[(size_t)k * SEQ + c];
    Wt[c][k] = gW[(size_t)k * HDIM + c];
#endif
  }
}

__global__ __launch_bounds__(128)
void qkv_proj_wmma(const float* __restrict__ X,
                   const float* __restrict__ Wq, const float* __restrict__ bq,
                   const float* __restrict__ Wk, const float* __restrict__ bk,
                   const float* __restrict__ Wv, const float* __restrict__ bv,
                   float* __restrict__ Qo, float* __restrict__ Ko,
                   float* __restrict__ Vo)
{
  // [buf][s][k] and [buf][n][k]; stride 36 floats = 144B (16B aligned rows)
  __shared__ __align__(16) float Axt[2][64][36];
  __shared__ __align__(16) float Wxt[2][64][36];

  const int tid  = threadIdx.x;        // 0..127
  const int lane = tid & 31;
  const int w    = tid >> 5;           // wave id 0..3 -> s stripe
  const int half = lane >> 4;
  const int mn   = lane & 15;
  const int s0   = blockIdx.x * 64;
  const int h    = blockIdx.y;
  const int b    = blockIdx.z / 3;
  const int mat  = blockIdx.z % 3;     // 0=Q 1=K 2=V

  const float* W    = (mat == 0) ? Wq : (mat == 1) ? Wk : Wv;
  const float* bias = (mat == 0) ? bq : (mat == 1) ? bk : bv;
  float*       dst  = (mat == 0) ? Qo : (mat == 1) ? Ko : Vo;
  const float  osc  = (mat == 2) ? 1.0f : INV_SCALE;

  const float* Xb = X + (size_t)b * EMBED * SEQ + s0;      // X[b,e,s0+..]
  const float* Wh = W + (size_t)h * EMBED * HDIM;          // W[h,e,d]

  v8f acc[4] = {};

  stage_slab(tid, Xb, Wh, Axt[0], Wxt[0]);                 // preload k0=0
  for (int kstep = 0; kstep < 16; ++kstep) {
    const int buf = kstep & 1;
    if (kstep + 1 < 16)
      stage_slab(tid, Xb + (size_t)(kstep + 1) * 32 * SEQ,
                 Wh + (size_t)(kstep + 1) * 32 * HDIM,
                 Axt[buf ^ 1], Wxt[buf ^ 1]);
#if HAVE_ASYNC
    if (kstep + 1 < 16) __builtin_amdgcn_s_wait_asynccnt(32);
    else                __builtin_amdgcn_s_wait_asynccnt(0);
#endif
    __syncthreads();

    // A fragment: rows M=mn of this wave's stripe; K runs are contiguous.
    const float* arow = &Axt[buf][w * 16 + mn][0];
    const v16h afrag = frag_from_f32(arow + half * 8, arow + 16 + half * 8);
#pragma unroll
    for (int nt = 0; nt < 4; ++nt) {
      const float* brow = &Wxt[buf][nt * 16 + mn][0];
      const v16h bfrag = frag_from_f32(brow + half * 8, brow + 16 + half * 8);
      acc[nt] = __builtin_amdgcn_wmma_f32_16x16x32_f16(
          false, afrag, false, bfrag, (short)0, acc[nt], false, false);
    }
    __syncthreads();
  }

  // C/D layout: VGPR r -> M = r + 8*half, N = mn
#pragma unroll
  for (int nt = 0; nt < 4; ++nt) {
    const float bb = bias[h * HDIM + nt * 16 + mn];
#pragma unroll
    for (int r = 0; r < 8; ++r) {
      const int srow = s0 + w * 16 + r + (half << 3);
      dst[(((size_t)b * NHEAD + h) * SEQ + srow) * HDIM + nt * 16 + mn] =
          (acc[nt][r] + bb) * osc;
    }
  }
}

// ---------------------------------------------------------------------------
// 8x8 Newton-Schulz pseudo-inverse (6 iterations), 64 threads cooperate.
// ---------------------------------------------------------------------------
__device__ void newton_inv8(int t,
                            float (*Gm)[9], float* dGv,
                            float (*Km)[9], float (*KV)[9],
                            float (*T1)[9], float (*T2)[9],
                            float (*Vout)[9],
                            float* rs8, float* cs8, float* sden)
{
  const int gi = t >> 3, gj = t & 7;
  Km[gi][gj] = Gm[gi][gj] / dGv[gi];
  __syncthreads();
  if (t < 8) {
    float r = 0.f, c = 0.f;
    for (int j = 0; j < 8; ++j) { r += fabsf(Km[t][j]); c += fabsf(Km[j][t]); }
    rs8[t] = r; cs8[t] = c;
  }
  __syncthreads();
  if (t == 0) {
    float mr = 0.f, mc = 0.f;
    for (int j = 0; j < 8; ++j) { mr = fmaxf(mr, rs8[j]); mc = fmaxf(mc, cs8[j]); }
    *sden = 1.0f / (mr * mc);
  }
  __syncthreads();
  Vout[gi][gj] = Km[gj][gi] * (*sden);
  __syncthreads();
  for (int it = 0; it < 6; ++it) {
    float a = 0.f;
    for (int l = 0; l < 8; ++l) a = fmaf(Km[gi][l], Vout[l][gj], a);
    KV[gi][gj] = a;
    __syncthreads();
    T1[gi][gj] = ((gi == gj) ? 7.0f : 0.0f) - KV[gi][gj];
    __syncthreads();
    float a2 = 0.f;
    for (int l = 0; l < 8; ++l) a2 = fmaf(KV[gi][l], T1[l][gj], a2);
    T2[gi][gj] = ((gi == gj) ? 15.0f : 0.0f) - a2;
    __syncthreads();
    float a3 = 0.f;
    for (int l = 0; l < 8; ++l) a3 = fmaf(KV[gi][l], T2[l][gj], a3);
    T1[gi][gj] = ((gi == gj) ? 13.0f : 0.0f) - a3;
    __syncthreads();
    float a4 = 0.f;
    for (int l = 0; l < 8; ++l) a4 = fmaf(Vout[gi][l], T1[l][gj], a4);
    T2[gi][gj] = 0.25f * a4;
    __syncthreads();
    Vout[gi][gj] = T2[gi][gj];
    __syncthreads();
  }
}

// ---------------------------------------------------------------------------
// Kernel 2: per-(b,h) Nystrom init + 960-step sequential scan.
// One block per (b,h); 64 threads; whole sliding state lives in LDS.
// Per-step Q/K/V row fetches use async global->LDS (latency-critical path).
// ---------------------------------------------------------------------------
__global__ __launch_bounds__(64)
void nystrom_scan(const float* __restrict__ Qg_, const float* __restrict__ Kg_,
                  const float* __restrict__ Vg_, float* __restrict__ out)
{
  __shared__ float Qr[WIN][65], Kr[WIN][65], Vr[WIN][65];   // ring buffers
  __shared__ float Qt[NLM][65], Kt[NLM][65];                // landmarks
  __shared__ float Delta[NLM][65];
  __shared__ float Beta[WIN][9], BGm[WIN][9];
  __shared__ float Gamma[NLM][9], GD[NLM][9];
  __shared__ float sKm[NLM][9], sKV[NLM][9], sT1[NLM][9], sT2[NLM][9];
  __shared__ float dB[WIN], dG[NLM], dD[NLM];
  __shared__ float qn[WIN], kn[WIN], vnn[WIN];
  __shared__ float qtn[WIN], ktn[WIN];
  __shared__ float wS[WIN];
  __shared__ float BnS[NLM], DnS[NLM], EoS[NLM];
  __shared__ float rs8[NLM], cs8[NLM];
  __shared__ float sden, sdBn;

  const int t  = threadIdx.x;          // 0..63
  const int bh = blockIdx.x;           // b*8 + h
  const int b  = bh >> 3, h = bh & 7;
  const int gi = t >> 3, gj = t & 7;
  const float* Qg = Qg_ + (size_t)bh * SEQ * HDIM;
  const float* Kg = Kg_ + (size_t)bh * SEQ * HDIM;
  const float* Vg = Vg_ + (size_t)bh * SEQ * HDIM;

  // ---------------- init: first window ----------------
  for (int r = 0; r < WIN; ++r) {
    Qr[r][t] = Qg[r * HDIM + t];
    Kr[r][t] = Kg[r * HDIM + t];
    Vr[r][t] = Vg[r * HDIM + t];
  }
  __syncthreads();
  for (int l = 0; l < NLM; ++l) {      // landmark means (col t)
    float sq = 0.f, sk = 0.f;
    for (int rr = 0; rr < 8; ++rr) { sq += Qr[l * 8 + rr][t]; sk += Kr[l * 8 + rr][t]; }
    Qt[l][t] = sq * 0.125f; Kt[l][t] = sk * 0.125f;
  }
  __syncthreads();
  {                                    // Beta row t + dB
    float s = 0.f;
    for (int j = 0; j < NLM; ++j) {
      float e = __expf(dot64(&Qr[t][0], &Kt[j][0]));
      Beta[t][j] = e; s += e;
    }
    dB[t] = s;
  }
  Gamma[gi][gj] = __expf(dot64(&Qt[gi][0], &Kt[gj][0]));
  for (int l = 0; l < NLM; ++l)        // Delta col t
    Delta[l][t] = __expf(dot64(&Qt[l][0], &Kr[t][0]));
  __syncthreads();
  if (t < NLM) {
    float sg = 0.f, sd = 0.f;
    for (int j = 0; j < NLM; ++j) sg += Gamma[t][j];
    for (int j = 0; j < WIN; ++j) sd += Delta[t][j];
    dG[t] = sg; dD[t] = sd;
  }
  __syncthreads();
  newton_inv8(t, Gamma, dG, sKm, sKV, sT1, sT2, GD, rs8, cs8, &sden);
  {                                    // BG row t
    const float invb = 1.0f / dB[t];
    for (int j = 0; j < NLM; ++j) {
      float a = 0.f;
      for (int l = 0; l < NLM; ++l) a = fmaf(Beta[t][l] * invb, GD[l][j], a);
      BGm[t][j] = a;
    }
  }
  __syncthreads();
  {                                    // attn_first, DV column t in regs
    float dv[NLM];
    for (int l = 0; l < NLM; ++l) {
      float a = 0.f;
      for (int r = 0; r < WIN; ++r) a = fmaf(Delta[l][r], Vr[r][t], a);
      dv[l] = a / dD[l];
    }
    for (int r = 0; r < WIN; ++r) {
      float a = 0.f;
      for (int l = 0; l < NLM; ++l) a = fmaf(BGm[r][l], dv[l], a);
      out[((size_t)b * SEQ + r) * (NHEAD * HDIM) + h * HDIM + t] = a;
    }
  }
  __syncthreads();

  // ---------------- scan: i = 64 .. 1023 ----------------
  for (int i = WIN; i < SEQ; ++i) {
    const int pos = i & 63;            // ring slot holding row i-64 (oldest)
#if HAVE_ASYNC
    __builtin_amdgcn_global_load_async_to_lds_b32(
        AS1G(Qg + (size_t)i * HDIM + t), AS3L(&qn[t]), 0, 0);
    __builtin_amdgcn_global_load_async_to_lds_b32(
        AS1G(Kg + (size_t)i * HDIM + t), AS3L(&kn[t]), 0, 0);
    __builtin_amdgcn_global_load_async_to_lds_b32(
        AS1G(Vg + (size_t)i * HDIM + t), AS3L(&vnn[t]), 0, 0);
#else
    qn[t]  = Qg[(size_t)i * HDIM + t];
    kn[t]  = Kg[(size_t)i * HDIM + t];
    vnn[t] = Vg[(size_t)i * HDIM + t];
#endif
    if (i + 1 < SEQ) {                 // prefetch next step's rows into L2/L0
      __builtin_prefetch(&Qg[(size_t)(i + 1) * HDIM + t], 0, 1);
      __builtin_prefetch(&Kg[(size_t)(i + 1) * HDIM + t], 0, 1);
      __builtin_prefetch(&Vg[(size_t)(i + 1) * HDIM + t], 0, 1);
    }
#if HAVE_ASYNC
    __builtin_amdgcn_s_wait_asynccnt(0);
#endif
    __syncthreads();

    if ((i & 7) == 7) {
      // ---------------- landmark branch ----------------
      {
        float sq = 0.f, sk = 0.f;
        for (int r = 0; r < WIN; ++r) { sq += Qr[r][t]; sk += Kr[r][t]; }
        qtn[t] = sq * (1.0f / 64.0f); ktn[t] = sk * (1.0f / 64.0f);
      }
      __syncthreads();
      float nBeta[NLM] = {}, nDelta[NLM] = {}, nQt[NLM] = {}, nKt[NLM] = {};
      float ndB = 0.f, nGam = 0.f, ndG = 0.f, ndD = 0.f;
      if (t < 63) {
        const float* qm = &Qr[(pos + 1 + t) & 63][0];   // Qm[t] = Q row i-63+t
        const float* km = &Kr[(pos + 1 + t) & 63][0];   // Km[t]
        for (int j = 0; j < 7; ++j) nBeta[j] = Beta[t + 1][j + 1];
        const float eb = __expf(dot64(qm, &ktn[0]));
        nBeta[7] = eb;
        ndB = dB[t + 1] - __expf(dot64(qm, &Kt[0][0])) + eb;   // k_told = old Kt[0]
        for (int l = 0; l < 7; ++l) nDelta[l] = Delta[l + 1][t + 1];
        nDelta[7] = __expf(dot64(&qtn[0], km));
      } else {
        float s = 0.f;
        for (int j = 0; j < 7; ++j) { nBeta[j] = __expf(dot64(&qn[0], &Kt[j + 1][0])); s += nBeta[j]; }
        nBeta[7] = __expf(dot64(&qn[0], &ktn[0]));
        ndB = s + nBeta[7];
        for (int l = 0; l < 7; ++l) nDelta[l] = __expf(dot64(&Qt[l + 1][0], &kn[0]));
        nDelta[7] = __expf(dot64(&qtn[0], &kn[0]));
      }
      if (gi < 7 && gj < 7)      nGam = Gamma[gi + 1][gj + 1];
      else if (gi < 7)           nGam = __expf(dot64(&Qt[gi + 1][0], &ktn[0]));
      else if (gj < 7)           nGam = __expf(dot64(&qtn[0], &Kt[gj + 1][0]));
      else                       nGam = __expf(dot64(&qtn[0], &ktn[0]));
      if (t < 7) {
        ndG = dG[t + 1] - __expf(dot64(&Qt[t + 1][0], &Kt[0][0]))
                        + __expf(dot64(&Qt[t + 1][0], &ktn[0]));
        // NOTE: reference's landmark branch uses Qw[...,0:1] (row i-64 of Q!)
        ndD = dD[t + 1] - __expf(dot64(&Qt[t + 1][0], &Qr[pos][0]))
                        + __expf(dot64(&Qt[t + 1][0], &kn[0]));
      }
      for (int l = 0; l < NLM; ++l) {
        nQt[l] = (l < 7) ? Qt[l + 1][t] : qtn[t];
        nKt[l] = (l < 7) ? Kt[l + 1][t] : ktn[t];
      }
      __syncthreads();
      // commit
      for (int j = 0; j < NLM; ++j) Beta[t][j] = nBeta[j];
      dB[t] = ndB;
      for (int l = 0; l < NLM; ++l) Delta[l][t] = nDelta[l];
      Gamma[gi][gj] = nGam;
      if (t < 7) { dG[t] = ndG; dD[t] = ndD; }
      for (int l = 0; l < NLM; ++l) { Qt[l][t] = nQt[l]; Kt[l][t] = nKt[l]; }
      __syncthreads();
      if (t == 0) {
        float sg = 0.f;
        for (int j = 0; j < NLM; ++j) sg += Gamma[7][j];
        dG[7] = sg;
        float sd = 0.f;                       // dD_new = sum(exp(q_tn.Km)) + exp(q_tn.k_tn)
        for (int j = 0; j < 63; ++j) sd += Delta[7][j];
        dD[7] = sd + Gamma[7][7];
      }
      __syncthreads();
      newton_inv8(t, Gamma, dG, sKm, sKV, sT1, sT2, GD, rs8, cs8, &sden);
      {
        const float invb = 1.0f / dB[t];
        for (int j = 0; j < NLM; ++j) {
          float a = 0.f;
          for (int l = 0; l < NLM; ++l) a = fmaf(Beta[t][l] * invb, GD[l][j], a);
          BGm[t][j] = a;
        }
      }
      __syncthreads();
    } else {
      // ---------------- shift branch ----------------
      if (t < 8)        BnS[t]      = __expf(dot64(&qn[0], &Kt[t][0]));
      else if (t < 16)  DnS[t - 8]  = __expf(dot64(&Qt[t - 8][0], &kn[0]));
      else if (t < 24)  EoS[t - 16] = __expf(dot64(&Qt[t - 16][0], &Kr[pos][0]));
      __syncthreads();
      float rBeta[NLM] = {}, rBG[NLM] = {}, rDelta[NLM] = {};
      float rdB = 0.f, rdD = 0.f;
      if (t == 0) {
        float s = 0.f;
        for (int j = 0; j < NLM; ++j) s += BnS[j];
        sdBn = s;
      }
      if (t < 63) {
        for (int j = 0; j < NLM; ++j) { rBeta[j] = Beta[t + 1][j]; rBG[j] = BGm[t + 1][j]; }
        rdB = dB[t + 1];
      } else {
        for (int j = 0; j < NLM; ++j) rBeta[j] = BnS[j];
      }
      for (int l = 0; l < NLM; ++l)
        rDelta[l] = (t < 63) ? Delta[l][t + 1] : DnS[l];
      if (t < 8) rdD = dD[t] - EoS[t] + DnS[t];
      __syncthreads();
      for (int j = 0; j < NLM; ++j) Beta[t][j] = rBeta[j];
      for (int l = 0; l < NLM; ++l) Delta[l][t] = rDelta[l];
      if (t < 63) {
        for (int j = 0; j < NLM; ++j) BGm[t][j] = rBG[j];
        dB[t] = rdB;
      } else {
        const float ib = 1.0f / sdBn;
        for (int j = 0; j < NLM; ++j) {
          float a = 0.f;
          for (int l = 0; l < NLM; ++l) a = fmaf(BnS[l] * ib, GD[l][j], a);
          BGm[63][j] = a;
        }
        dB[63] = sdBn;
      }
      if (t < 8) dD[t] = rdD;
      __syncthreads();
    }

    // ring update (old slot contents already consumed above)
    Qr[pos][t] = qn[t]; Kr[pos][t] = kn[t]; Vr[pos][t] = vnn[t];
    __syncthreads();

    // res = row0(BG2) @ (Delta2/dD2 @ Vw[1:]) fused via w-vector
    {
      float a = 0.f;
      for (int l = 0; l < NLM; ++l) a = fmaf(BGm[0][l] / dD[l], Delta[l][t], a);
      wS[(pos + 1 + t) & 63] = a;      // column t' -> ring slot
    }
    __syncthreads();
    {
      float a = 0.f;
      for (int r = 0; r < WIN; ++r) a = fmaf(wS[r], Vr[r][t], a);
      out[((size_t)b * SEQ + i) * (NHEAD * HDIM) + h * HDIM + t] = a;
    }
    __syncthreads();
  }
}

extern "C" void kernel_launch(void* const* d_in, const int* in_sizes, int n_in,
                              void* d_out, int out_size, void* d_ws, size_t ws_size,
                              hipStream_t stream) {
  (void)in_sizes; (void)n_in; (void)out_size; (void)ws_size;
  const float* X  = (const float*)d_in[0];
  const float* Wq = (const float*)d_in[1];
  const float* bq = (const float*)d_in[2];
  const float* Wk = (const float*)d_in[3];
  const float* bk = (const float*)d_in[4];
  const float* Wv = (const float*)d_in[5];
  const float* bv = (const float*)d_in[6];
  float* out = (float*)d_out;

  const size_t per = (size_t)NBATCH * NHEAD * SEQ * HDIM;  // 8,388,608 floats
  float* wsQ = (float*)d_ws;
  float* wsK = wsQ + per;
  float* wsV = wsK + per;

  dim3 g1(SEQ / 64, NHEAD, NBATCH * 3);
  qkv_proj_wmma<<<g1, 128, 0, stream>>>(X, Wq, bq, Wk, bk, Wv, bv, wsQ, wsK, wsV);
  nystrom_scan<<<dim3(NBATCH * NHEAD), 64, 0, stream>>>(wsQ, wsK, wsV, out);
}
